// DiscoveryEngineModel_2843268350307
// MI455X (gfx1250) — matrix operand order repaired and checked by reading, not compiled, not run on hardware
//
#include <hip/hip_runtime.h>
#include <math.h>

typedef __attribute__((ext_vector_type(16))) _Float16 v16h;
typedef __attribute__((ext_vector_type(8)))  float    v8f;

#define HID  128
#define KPAD 288   // 258/257 padded up to 9 chunks of K=32
#define MB   32    // rows (edges or nodes) per block

// ---------------- WMMA fragment helpers (layouts per CDNA5 ISA) ----

// A-matrix 16x32 f16 from LDS. Lane l: row m = m0 + (l&15); lanes 0-15 cover
// K = k0+{0..7, 16..23}, lanes 16-31 cover K = k0+{8..15, 24..31}.
__device__ __forceinline__ v16h ldsA(const _Float16* base, int lda, int m0, int k0) {
  int lane = threadIdx.x & 31;
  const _Float16* r = base + (m0 + (lane & 15)) * lda + k0 + ((lane & 16) ? 8 : 0);
  v16h a;
#pragma unroll
  for (int j = 0; j < 8; ++j) a[j] = r[j];
#pragma unroll
  for (int j = 0; j < 8; ++j) a[8 + j] = r[16 + j];
  return a;
}

// B-matrix 32x16 f16 from global weights stored [n_out][K] row-major.
// Lane l: column n = n0 + (l&15); lanes 0-15 hold K=k0+{0..7, 16..23}, lanes 16-31 K=k0+{8..15, 24..31}.
__device__ __forceinline__ v16h ldB(const _Float16* w, int ldb, int n0, int k0) {
  int lane = threadIdx.x & 31;
  const _Float16* r = w + (long)(n0 + (lane & 15)) * ldb + k0 + ((lane & 16) ? 8 : 0);
  v16h b;
#pragma unroll
  for (int j = 0; j < 8; ++j) b[j] = r[j];
#pragma unroll
  for (int j = 0; j < 8; ++j) b[8 + j] = r[16 + j];
  return b;
}

// One 16xKTOT * KTOTx32 tile pair: A loaded once per K step, applied to two N-tiles.
template <int KTOT>
__device__ __forceinline__ void gemm2(const _Float16* A, int lda, const _Float16* W, int ldb,
                                      int m0, int n0, v8f& c0, v8f& c1) {
#pragma unroll
  for (int k = 0; k < KTOT; k += 32) {
    v16h a  = ldsA(A, lda, m0, k);
    v16h b0 = ldB(W, ldb, n0, k);
    v16h b1 = ldB(W, ldb, n0 + 16, k);
    c0 = __builtin_amdgcn_wmma_f32_16x16x32_f16(false, a, false, b0, (short)0, c0, false, false);
    c1 = __builtin_amdgcn_wmma_f32_16x16x32_f16(false, a, false, b1, (short)0, c1, false, false);
  }
}

// C/D fragment 16x16 f32: lane l owns column n = n0+(l&15), rows m0 + ((l&16)?8:0) + r.
__device__ __forceinline__ void store_act(_Float16* out, int ldo, int m0, int n0,
                                          const float* __restrict__ bias, v8f c, bool act) {
  int lane = threadIdx.x & 31;
  int n  = n0 + (lane & 15);
  int mb = (lane & 16) ? 8 : 0;
  float bs = bias[n];
#pragma unroll
  for (int r = 0; r < 8; ++r) {
    float v = c[r] + bs;
    if (act) v = v / (1.f + __expf(-v));   // SiLU
    out[(m0 + mb + r) * ldo + n] = (_Float16)v;
  }
}

// ---------------- prep kernels -----------------------------------------------

__global__ void zero_kernel(float* p, long n) {
  long i = (long)blockIdx.x * blockDim.x + threadIdx.x;
  if (i < n) p[i] = 0.f;
}

// f32 [rows][ksrc] -> f16 [rows][kdst], zero-padded tail columns
__global__ void cvt_kernel(const float* __restrict__ src, _Float16* __restrict__ dst,
                           int rows, int ksrc, int kdst) {
  int i = blockIdx.x * blockDim.x + threadIdx.x;
  if (i < rows * kdst) {
    int r = i / kdst, k = i - r * kdst;
    dst[i] = (k < ksrc) ? (_Float16)src[r * ksrc + k] : (_Float16)0.f;
  }
}

// ---------------- fused edge kernel ------------------------------------------
// 256 threads = 8 waves; 32 edges per block. Wave w: M-tile (w>>2)*16,
// N-tiles (w&3)*32 and (w&3)*32+16 -> 8 waves cover the full 32x128 tile.

__global__ __launch_bounds__(256) void egnn_edge_kernel(
    const float* __restrict__ x, const float* __restrict__ pos, const float* __restrict__ vel,
    const int* __restrict__ ei, int E_,
    const _Float16* __restrict__ We1h, const float* __restrict__ be1,
    const _Float16* __restrict__ We2h, const float* __restrict__ be2,
    const _Float16* __restrict__ We3h, const float* __restrict__ be3,
    const _Float16* __restrict__ Wv1h, const float* __restrict__ bv1,
    const float* __restrict__ Wv2,  const float* __restrict__ bv2,
    float* __restrict__ mh_agg, float* __restrict__ mv_agg)
{
  __shared__ _Float16 sTmp[MB * KPAD];   // edge features (reused as layer-2 out)
  __shared__ _Float16 sH[MB * HID];      // phi_e layer-1 out
  __shared__ _Float16 sV[MB * HID];      // phi_v layer-1 out
  __shared__ float sRel[MB * 2];
  __shared__ int sSrc[MB];
  __shared__ int sDst[MB];

  const int tid = threadIdx.x;
  const int e0  = blockIdx.x * MB;

  // per-edge geometry (threads 0..31, coalesced index loads)
  if (tid < MB) {
    int e = e0 + tid;
    bool ok = e < E_;
    int s = ok ? ei[e]      : 0;
    int d = ok ? ei[E_ + e] : 0;
    sSrc[tid] = s; sDst[tid] = d;
    float rpx = pos[2 * s]     - pos[2 * d];
    float rpy = pos[2 * s + 1] - pos[2 * d + 1];
    float rvx = vel[2 * s]     - vel[2 * d];
    float rvy = vel[2 * s + 1] - vel[2 * d + 1];
    if (!ok) { rpx = rpy = rvx = rvy = 0.f; }
    sRel[2 * tid] = rpx; sRel[2 * tid + 1] = rpy;
    _Float16* row = sTmp + tid * KPAD;
    row[256] = (_Float16)(rpx * rpx + rpy * rpy);   // dist_sq
    row[257] = (_Float16)(rvx * rpx + rvy * rpy);   // dot(v, r)
#pragma unroll
    for (int k = 258; k < KPAD; ++k) row[k] = (_Float16)0.f;
  }
  __syncthreads();

  // gather [x[dst] | x[src]] -> sTmp cols 0..255 (coalesced 128-wide rows, L2-hot)
  for (int idx = tid; idx < MB * 256; idx += 256) {
    int m = idx >> 8, c = idx & 255;
    int node = (c < 128) ? sDst[m] : sSrc[m];
    sTmp[m * KPAD + c] = (_Float16)x[(long)node * 128 + (c & 127)];
  }
  __syncthreads();

  const int w  = tid >> 5;
  const int m0 = (w >> 2) * 16;
  const int n0 = (w & 3) * 32;

  { // layer 1 of phi_e and phi_v (K = 288 -> 9 WMMAs per N-tile each)
    v8f ce0 = {}, ce1 = {}, cv0 = {}, cv1 = {};
    gemm2<KPAD>(sTmp, KPAD, We1h, KPAD, m0, n0, ce0, ce1);
    gemm2<KPAD>(sTmp, KPAD, Wv1h, KPAD, m0, n0, cv0, cv1);
    store_act(sH, HID, m0, n0,      be1, ce0, true);
    store_act(sH, HID, m0, n0 + 16, be1, ce1, true);
    store_act(sV, HID, m0, n0,      bv1, cv0, true);
    store_act(sV, HID, m0, n0 + 16, bv1, cv1, true);
  }
  __syncthreads();

  // phi_v head (128->1) + m_v scatter, one edge per lane
  if (tid < MB && (e0 + tid) < E_) {
    float acc = bv2[0];
    const _Float16* vr = sV + tid * HID;
#pragma unroll 8
    for (int k = 0; k < HID; ++k) acc += (float)vr[k] * Wv2[k];
    int d = sDst[tid];
    unsafeAtomicAdd(&mv_agg[2 * d],     acc * sRel[2 * tid]);
    unsafeAtomicAdd(&mv_agg[2 * d + 1], acc * sRel[2 * tid + 1]);
  }

  _Float16* sH2 = sTmp;   // reuse input tile buffer for layer-2 output
  { // phi_e layer 2
    v8f c0 = {}, c1 = {};
    gemm2<HID>(sH, HID, We2h, HID, m0, n0, c0, c1);
    store_act(sH2, HID, m0, n0,      be2, c0, true);
    store_act(sH2, HID, m0, n0 + 16, be2, c1, true);
  }
  __syncthreads();

  { // phi_e layer 3 + segment-sum scatter straight from C fragments
    v8f c0 = {}, c1 = {};
    gemm2<HID>(sH2, HID, We3h, HID, m0, n0, c0, c1);
    int lane = tid & 31;
    int n  = n0 + (lane & 15);
    int mb = (lane & 16) ? 8 : 0;
    float b0 = be3[n], b1 = be3[n + 16];
#pragma unroll
    for (int r = 0; r < 8; ++r) {
      int m = m0 + mb + r;
      if (e0 + m < E_) {
        long d = sDst[m];
        unsafeAtomicAdd(&mh_agg[d * 128 + n],      c0[r] + b0);
        unsafeAtomicAdd(&mh_agg[d * 128 + n + 16], c1[r] + b1);
      }
    }
  }
}

// ---------------- fused node kernel ------------------------------------------

__global__ __launch_bounds__(256) void egnn_node_kernel(
    const float* __restrict__ x, const float* __restrict__ mh_agg, const float* __restrict__ mv_agg,
    const _Float16* __restrict__ Wh1h, const float* __restrict__ bh1,
    const _Float16* __restrict__ Wh2h, const float* __restrict__ bh2,
    float* __restrict__ out, int N_)
{
  __shared__ _Float16 sIn[MB * KPAD];
  __shared__ _Float16 sHid[MB * HID];

  const int tid  = threadIdx.x;
  const int base = blockIdx.x * MB;

  if (tid < MB) {
    int i = base + tid; if (i >= N_) i = N_ - 1;
    float mx = mv_agg[2 * i], my = mv_agg[2 * i + 1];
    float nrm = sqrtf(fmaxf(mx * mx + my * my, 1e-24f));
    _Float16* row = sIn + tid * KPAD;
    row[256] = (_Float16)nrm;
#pragma unroll
    for (int k = 257; k < KPAD; ++k) row[k] = (_Float16)0.f;
  }
  for (int idx = tid; idx < MB * 256; idx += 256) {
    int m = idx >> 8, c = idx & 255;
    int i = base + m; if (i >= N_) i = N_ - 1;
    float v = (c < 128) ? x[(long)i * 128 + c] : mh_agg[(long)i * 128 + (c - 128)];
    sIn[m * KPAD + c] = (_Float16)v;
  }
  __syncthreads();

  const int w  = tid >> 5;
  const int m0 = (w >> 2) * 16;
  const int n0 = (w & 3) * 32;

  { // phi_h layer 1
    v8f c0 = {}, c1 = {};
    gemm2<KPAD>(sIn, KPAD, Wh1h, KPAD, m0, n0, c0, c1);
    store_act(sHid, HID, m0, n0,      bh1, c0, true);
    store_act(sHid, HID, m0, n0 + 16, bh1, c1, true);
  }
  __syncthreads();

  { // phi_h layer 2 + residual store
    v8f c0 = {}, c1 = {};
    gemm2<HID>(sHid, HID, Wh2h, HID, m0, n0, c0, c1);
    int lane = tid & 31;
    int n  = n0 + (lane & 15);
    int mb = (lane & 16) ? 8 : 0;
    float b0 = bh2[n], b1 = bh2[n + 16];
#pragma unroll
    for (int r = 0; r < 8; ++r) {
      int m = m0 + mb + r;
      int i = base + m;
      if (i < N_) {
        out[(long)i * 128 + n]      = x[(long)i * 128 + n]      + c0[r] + b0;
        out[(long)i * 128 + n + 16] = x[(long)i * 128 + n + 16] + c1[r] + b1;
      }
    }
  }
}

// ---------------- host launch ------------------------------------------------

extern "C" void kernel_launch(void* const* d_in, const int* in_sizes, int n_in,
                              void* d_out, int out_size, void* d_ws, size_t ws_size,
                              hipStream_t stream) {
  const float* x   = (const float*)d_in[0];
  const float* pos = (const float*)d_in[1];
  const float* vel = (const float*)d_in[2];
  const int*   ei  = (const int*)d_in[3];
  const float* We1 = (const float*)d_in[4];
  const float* be1 = (const float*)d_in[5];
  const float* We2 = (const float*)d_in[6];
  const float* be2 = (const float*)d_in[7];
  const float* We3 = (const float*)d_in[8];
  const float* be3 = (const float*)d_in[9];
  const float* Wv1 = (const float*)d_in[10];
  const float* bv1 = (const float*)d_in[11];
  const float* Wv2 = (const float*)d_in[12];
  const float* bv2 = (const float*)d_in[13];
  const float* Wh1 = (const float*)d_in[14];
  const float* bh1 = (const float*)d_in[15];
  const float* Wh2 = (const float*)d_in[16];
  const float* bh2 = (const float*)d_in[17];

  const int N = in_sizes[0] / 128;
  const int E = in_sizes[3] / 2;

  char* ws = (char*)d_ws;
  size_t off = 0;
  float* mh_agg = (float*)(ws + off); off += (size_t)N * 128 * sizeof(float);
  float* mv_agg = (float*)(ws + off); off += (size_t)N * 2 * sizeof(float);
  off = (off + 255) & ~(size_t)255;
  _Float16* We1h = (_Float16*)(ws + off); off += (size_t)128 * KPAD * 2;
  _Float16* Wv1h = (_Float16*)(ws + off); off += (size_t)128 * KPAD * 2;
  _Float16* Wh1h = (_Float16*)(ws + off); off += (size_t)128 * KPAD * 2;
  _Float16* We2h = (_Float16*)(ws + off); off += (size_t)128 * 128 * 2;
  _Float16* We3h = (_Float16*)(ws + off); off += (size_t)128 * 128 * 2;
  _Float16* Wh2h = (_Float16*)(ws + off); off += (size_t)128 * 128 * 2;

  // zero aggregation buffers (mh_agg and mv_agg are contiguous: N*130 floats)
  long nz = (long)N * 130;
  zero_kernel<<<(int)((nz + 255) / 256), 256, 0, stream>>>(mh_agg, nz);

  // one-time f16 weight conversion (tiny; stays L2-resident afterwards)
  cvt_kernel<<<(128 * KPAD + 255) / 256, 256, 0, stream>>>(We1, We1h, 128, 258, KPAD);
  cvt_kernel<<<(128 * KPAD + 255) / 256, 256, 0, stream>>>(Wv1, Wv1h, 128, 258, KPAD);
  cvt_kernel<<<(128 * KPAD + 255) / 256, 256, 0, stream>>>(Wh1, Wh1h, 128, 257, KPAD);
  cvt_kernel<<<(128 * 128 + 255) / 256, 256, 0, stream>>>(We2, We2h, 128, 128, 128);
  cvt_kernel<<<(128 * 128 + 255) / 256, 256, 0, stream>>>(We3, We3h, 128, 128, 128);
  cvt_kernel<<<(128 * 128 + 255) / 256, 256, 0, stream>>>(Wh2, Wh2h, 128, 128, 128);

  egnn_edge_kernel<<<(E + MB - 1) / MB, 256, 0, stream>>>(
      x, pos, vel, ei, E,
      We1h, be1, We2h, be2, We3h, be3, Wv1h, bv1, Wv2, bv2,
      mh_agg, mv_agg);

  egnn_node_kernel<<<(N + MB - 1) / MB, 256, 0, stream>>>(
      x, mh_agg, mv_agg, Wh1h, bh1, Wh2h, bh2, (float*)d_out, N);
}
